// MemoryGraph_22308060135897
// MI455X (gfx1250) — matrix operand order repaired and checked by reading, not compiled, not run on hardware
//
#include <hip/hip_runtime.h>
#include <hip/hip_bf16.h>

// ---------------------------------------------------------------------------
// Problem constants (from the reference)
// ---------------------------------------------------------------------------
#define BS_   8
#define T_    256
#define D_    2048
#define N_    8192
#define K_    64
#define G_    512
#define GS_   16
#define H_    64
#define MODIN_  69   // K + 5
#define MODOUT_ 66   // K + 2

typedef __attribute__((ext_vector_type(16))) _Float16 v16h;
typedef __attribute__((ext_vector_type(8)))  _Float16 v8h;
typedef __attribute__((ext_vector_type(8)))  float    v8f;

#if defined(__has_builtin)
#if __has_builtin(__builtin_amdgcn_global_load_async_to_lds_b32) && \
    __has_builtin(__builtin_amdgcn_s_wait_asynccnt)
#define HAVE_ASYNC_LDS 1
#endif
#endif

#ifdef HAVE_ASYNC_LDS
typedef __attribute__((address_space(1))) int gas_int;  // global addrspace
typedef __attribute__((address_space(3))) int las_int;  // LDS addrspace
#endif

// ---------------------------------------------------------------------------
// WMMA fragment builders from LDS, A and B in one contiguous K order (k = 16*(lane/16) + element).
// 16-bit A (16x32): lanes 0-15 -> M=lane,    halves 0-15 -> K=0..15
//                   lanes 16-31 -> M=lane-16, halves 0-15 -> K=16..31
//   => per lane: two K-contiguous 16B chunks (aligned) when A is row-major in LDS.
// 16-bit B (32x16): lanes 0-15 -> N=lane,  halves h -> K=h;
//                   lanes 16-31 -> N=lane-16, halves h -> K=16+h
//   => per lane: one K-contiguous 32B chunk when B is staged TRANSPOSED
//      (col-major, Wt[col][k]) -> a single aligned v16h (2x ds_load_b128).
// f32 C/D (16x16):  vgpr j, lanes 0-15 -> M=j, lanes 16-31 -> M=j+8; N=lane%16
// ---------------------------------------------------------------------------
__device__ __forceinline__ v16h load_a_frag(const _Float16* X, int ldx,
                                            int mbase, int kbase, int lane) {
    const int row = mbase + (lane & 15);
    const int hi  = (lane < 16) ? 0 : 16;
    const _Float16* p = X + (size_t)row * ldx + kbase + hi;
    v8h lo = *(const v8h*)(p);         // K = kbase+hi   .. +7   (16B aligned)
    v8h up = *(const v8h*)(p + 8);     // K = kbase+hi+8 .. +15  (16B aligned)
    return __builtin_shufflevector(lo, up, 0, 1, 2, 3, 4, 5, 6, 7,
                                           8, 9, 10, 11, 12, 13, 14, 15);
}

// Wt is stored transposed: Wt[col][k], row stride ldk (halves).
__device__ __forceinline__ v16h load_b_frag_t(const _Float16* Wt, int ldk,
                                              int kbase, int nbase, int lane) {
    const int col  = nbase + (lane & 15);
    const int koff = (lane < 16) ? 0 : 16;
    const _Float16* p = Wt + (size_t)col * ldk + kbase + koff;  // 32B aligned
    return *(const v16h*)p;            // 16 K-contiguous halves
}

// ---------------------------------------------------------------------------
// Phase 1: grouped MLP modulator. One block per group g (512 blocks, 128 thr
// = 4 waves). Per group:  X(128x69) -> tanh(X@W1 + b1) (128x64)
//                         -> @W2 + b2 (128x66) -> w_conn / decay / threshold
// All matmuls through v_wmma_f32_16x16x32_f16 (f16 in, f32 accumulate).
// ---------------------------------------------------------------------------
__global__ __launch_bounds__(128)
void modulator_kernel(const float* __restrict__ w1, const float* __restrict__ b1,
                      const float* __restrict__ w2, const float* __restrict__ b2,
                      const float* __restrict__ act0, const float* __restrict__ heb0,
                      const float* __restrict__ dec0, const float* __restrict__ thr0,
                      const float* __restrict__ nid,
                      float* __restrict__ w_conn, float* __restrict__ decay,
                      float* __restrict__ thresh) {
    __shared__ _Float16 Xs[128][96];   // X, row-major, K padded 69 -> 96
    __shared__ _Float16 W1t[64][96];   // W1 transposed: [h][k], k padded -> 96
    __shared__ _Float16 Hs[128][64];   // hidden, row-major
    __shared__ _Float16 W2t[80][64];   // W2 transposed: [o][k], o padded -> 80
    __shared__ float    b1s[64];
    __shared__ float    b2s[80];

    const int g = blockIdx.x, tid = threadIdx.x;

    // Stage inputs (f32 -> f16).
    for (int idx = tid; idx < 128 * 96; idx += 128) {
        const int r = idx / 96, c = idx % 96;
        const int bb = r >> 4, s = r & 15, n = g * GS_ + s;
        float v = 0.f;
        if      (c == 0)  v = act0[(size_t)bb * N_ + n];
        else if (c <= 64) v = heb0[((size_t)bb * N_ + n) * K_ + (c - 1)];
        else if (c == 65) v = dec0[(size_t)bb * N_ + n];
        else if (c == 66) v = thr0[(size_t)bb * N_ + n];
        else if (c == 67) v = nid[2 * n];
        else if (c == 68) v = nid[2 * n + 1];
        Xs[r][c] = (_Float16)v;
    }
    // W1 -> transposed LDS (coalesced global read, strided LDS write).
    for (int idx = tid; idx < 96 * 64; idx += 128) {
        const int k = idx >> 6, h = idx & 63;
        W1t[h][k] = (_Float16)((k < MODIN_) ? w1[((size_t)g * MODIN_ + k) * H_ + h] : 0.f);
    }
    for (int idx = tid; idx < 64 * 80; idx += 128) {
        const int k = idx / 80, o = idx % 80;
        W2t[o][k] = (_Float16)((o < MODOUT_) ? w2[((size_t)g * H_ + k) * MODOUT_ + o] : 0.f);
    }
    if (tid < 64) b1s[tid] = b1[(size_t)g * H_ + tid];
    if (tid < 80) b2s[tid] = (tid < MODOUT_) ? b2[(size_t)g * MODOUT_ + tid] : 0.f;
    __syncthreads();

    const int wave = tid >> 5, lane = tid & 31;

    // Each wave owns two 16-row tiles of the 128-row M dimension.
#pragma unroll
    for (int mi = 0; mi < 2; ++mi) {
        const int mbase = (wave * 2 + mi) * 16;

        // GEMM1: (16x96) @ (96x64), K in 3 steps of 32, 4 col tiles.
        for (int nt = 0; nt < 4; ++nt) {
            v8f acc = {};
#pragma unroll
            for (int ks = 0; ks < 3; ++ks) {
                v16h af = load_a_frag(&Xs[0][0], 96, mbase, ks * 32, lane);
                v16h bf = load_b_frag_t(&W1t[0][0], 96, ks * 32, nt * 16, lane);
                acc = __builtin_amdgcn_wmma_f32_16x16x32_f16(
                          false, af, false, bf, (short)0, acc, false, false);
            }
            const int col = nt * 16 + (lane & 15);
            const int rb  = mbase + ((lane < 16) ? 0 : 8);
#pragma unroll
            for (int j = 0; j < 8; ++j)
                Hs[rb + j][col] = (_Float16)tanhf(acc[j] + b1s[col]);
        }
        __syncthreads();

        // GEMM2: (16x64) @ (64x80), K in 2 steps of 32, 5 col tiles.
        for (int nt = 0; nt < 5; ++nt) {
            v8f acc = {};
#pragma unroll
            for (int ks = 0; ks < 2; ++ks) {
                v16h af = load_a_frag(&Hs[0][0], 64, mbase, ks * 32, lane);
                v16h bf = load_b_frag_t(&W2t[0][0], 64, ks * 32, nt * 16, lane);
                acc = __builtin_amdgcn_wmma_f32_16x16x32_f16(
                          false, af, false, bf, (short)0, acc, false, false);
            }
            const int col = nt * 16 + (lane & 15);
            const int rb  = mbase + ((lane < 16) ? 0 : 8);
#pragma unroll
            for (int j = 0; j < 8; ++j) {
                const int r = rb + j;
                const int bb = r >> 4, s = r & 15, n = g * GS_ + s;
                const float v = acc[j] + b2s[col];
                if (col < K_)
                    w_conn[((size_t)bb * N_ + n) * K_ + col] = v;
                else if (col == K_)
                    decay[(size_t)bb * N_ + n] = 1.f / (1.f + __expf(-v));
                else if (col == K_ + 1)
                    thresh[(size_t)bb * N_ + n] = v;
            }
        }
        __syncthreads();
    }
}

// ---------------------------------------------------------------------------
// Phase 2: 256-step recurrent scan. One block per batch (act fits in 32 KB
// LDS); 1024 threads, 8 neurons each. conn/w_conn re-read per step are
// L2-resident (192 MB L2 >> 18 MB working set). Hebbian update is dead code
// in the reference (never observable) and is omitted. Output mean over R=4
// is fused. Injection rows are double-buffered via async global->LDS copies.
// ---------------------------------------------------------------------------
__global__ __launch_bounds__(1024)
void scan_kernel(const float* __restrict__ cc, const int* __restrict__ conn,
                 const float* __restrict__ w_conn, const float* __restrict__ decay,
                 const float* __restrict__ thresh, const float* __restrict__ V0,
                 const float* __restrict__ act0, float* __restrict__ out) {
    __shared__ float act_s[N_];
#ifdef HAVE_ASYNC_LDS
    __shared__ float cc_s[2][D_];
#endif
    const int b = blockIdx.x, tid = threadIdx.x;
    const int n0 = tid * 8;      // 8 neurons per thread
    const int d0 = tid * 2;      // exactly 2 output lanes per thread (R=4)

    float V[8], dec[8], th[8];
#pragma unroll
    for (int j = 0; j < 8; ++j) {
        const size_t idx = (size_t)b * N_ + n0 + j;
        V[j]   = V0[idx];
        dec[j] = decay[idx];
        th[j]  = thresh[idx];
        act_s[n0 + j] = act0[idx];
    }
    const float* ccb = cc + (size_t)b * T_ * D_;

#ifdef HAVE_ASYNC_LDS
    __builtin_amdgcn_global_load_async_to_lds_b32(
        (gas_int*)(ccb + d0), (las_int*)&cc_s[0][d0], 0, 0);
    __builtin_amdgcn_global_load_async_to_lds_b32(
        (gas_int*)(ccb + d0 + 1), (las_int*)&cc_s[0][d0 + 1], 0, 0);
#endif
    __syncthreads();

    for (int t = 0; t < T_; ++t) {
#ifdef HAVE_ASYNC_LDS
        if (t + 1 < T_) {   // prefetch next injection row, wait for current
            const float* nxt = ccb + (size_t)(t + 1) * D_;
            const int nb = (t + 1) & 1;
            __builtin_amdgcn_global_load_async_to_lds_b32(
                (gas_int*)(nxt + d0), (las_int*)&cc_s[nb][d0], 0, 0);
            __builtin_amdgcn_global_load_async_to_lds_b32(
                (gas_int*)(nxt + d0 + 1), (las_int*)&cc_s[nb][d0 + 1], 0, 0);
            __builtin_amdgcn_s_wait_asynccnt(2);
        } else {
            __builtin_amdgcn_s_wait_asynccnt(0);
        }
        const float cc0 = cc_s[t & 1][d0];
        const float cc1 = cc_s[t & 1][d0 + 1];
#else
        const float cc0 = ccb[(size_t)t * D_ + d0];
        const float cc1 = ccb[(size_t)t * D_ + d0 + 1];
#endif

        float an[8];
#pragma unroll
        for (int j = 0; j < 8; ++j) {
            const int n = n0 + j;
            const int4*   ci = (const int4*)(conn + (size_t)n * K_);
            const float4* wv = (const float4*)(w_conn + ((size_t)b * N_ + n) * K_);
            float s = 0.f;
#pragma unroll 4
            for (int q = 0; q < 16; ++q) {      // 64 gathers, b128 loads
                const int4   i4 = ci[q];
                const float4 w4 = wv[q];
                s += act_s[i4.x] * w4.x + act_s[i4.y] * w4.y
                   + act_s[i4.z] * w4.z + act_s[i4.w] * w4.w;
            }
            const float recv = s + ((j < 4) ? cc0 : cc1);
            V[j]  = dec[j] * V[j] + (1.f - dec[j]) * recv;
            an[j] = 1.f / (1.f + __expf(th[j] - V[j]));   // sigmoid(V - th)
        }
        __syncthreads();                 // all gathers of old act done
#pragma unroll
        for (int j = 0; j < 8; ++j) act_s[n0 + j] = an[j];

        float* orow = out + ((size_t)b * T_ + t) * D_;
        orow[d0]     = (an[0] + an[1] + an[2] + an[3]) * 0.25f;
        orow[d0 + 1] = (an[4] + an[5] + an[6] + an[7]) * 0.25f;
        __syncthreads();                 // new act visible before next step
    }
}

// ---------------------------------------------------------------------------
extern "C" void kernel_launch(void* const* d_in, const int* in_sizes, int n_in,
                              void* d_out, int out_size, void* d_ws, size_t ws_size,
                              hipStream_t stream) {
    (void)in_sizes; (void)n_in; (void)out_size; (void)ws_size;
    const float* cc   = (const float*)d_in[0];
    const float* w1   = (const float*)d_in[1];
    const float* b1   = (const float*)d_in[2];
    const float* w2   = (const float*)d_in[3];
    const float* b2   = (const float*)d_in[4];
    const int*   conn = (const int*)d_in[5];
    const float* nid  = (const float*)d_in[6];
    const float* V0   = (const float*)d_in[7];
    const float* act0 = (const float*)d_in[8];
    const float* heb0 = (const float*)d_in[9];
    const float* dec0 = (const float*)d_in[10];
    const float* thr0 = (const float*)d_in[11];

    float* wconn = (float*)d_ws;                        // BS*N*K = 16 MB
    float* dcy   = wconn + (size_t)BS_ * N_ * K_;       // BS*N
    float* thr   = dcy   + (size_t)BS_ * N_;            // BS*N

    modulator_kernel<<<dim3(G_), dim3(128), 0, stream>>>(
        w1, b1, w2, b2, act0, heb0, dec0, thr0, nid, wconn, dcy, thr);
    scan_kernel<<<dim3(BS_), dim3(1024), 0, stream>>>(
        cc, conn, wconn, dcy, thr, V0, act0, (float*)d_out);
}
